// SwinTransformerBlock_37434934952506
// MI455X (gfx1250) — matrix operand order repaired and checked by reading, not compiled, not run on hardware
//
#include <hip/hip_runtime.h>
#include <hip/hip_bf16.h>
#include <math.h>

// ---------------- problem constants ----------------
#define BB    32
#define HH    56
#define CC    192
#define WS7   7
#define SS3   3
#define NH6   6
#define DH32  32
#define NWIN  64                 // (56/7)^2
#define NTOK  49                 // 7*7
#define MTOT  (BB*HH*HH)         // 100352 tokens
#define QKSCALE 0.17677669529663687f  // 32^-0.5

typedef __attribute__((ext_vector_type(8)))  float   v8f;
typedef __attribute__((ext_vector_type(16))) __bf16  v16bf;
typedef int v4i __attribute__((vector_size(16)));

#define AS1 __attribute__((address_space(1)))
#define AS3 __attribute__((address_space(3)))

#if __has_builtin(__builtin_amdgcn_global_load_async_to_lds_b128)
#define HAS_ASYNC_LDS 1
#else
#define HAS_ASYNC_LDS 0
#endif

__device__ __forceinline__ void async_wait0() {
#if HAS_ASYNC_LDS
#if __has_builtin(__builtin_amdgcn_s_wait_asynccnt)
  __builtin_amdgcn_s_wait_asynccnt(0);
#else
  asm volatile("s_wait_asynccnt 0" ::: "memory");
#endif
#endif
}

__device__ __forceinline__ unsigned short f2bf(float f) {
  union { float f; unsigned u; } v; v.f = f;
  unsigned r = v.u + 0x7FFFu + ((v.u >> 16) & 1u);
  return (unsigned short)(r >> 16);
}

__device__ __forceinline__ v8f vzero8() {
  v8f v;
  #pragma unroll
  for (int i = 0; i < 8; ++i) v[i] = 0.f;
  return v;
}

union Frag16 { v16bf v; uint4 q[2]; };

__device__ __forceinline__ int region3(int h) { return (h < 49) ? 0 : ((h < 53) ? 1 : 2); }

// ---------------- weight convert + transpose: W[K][N] f32 -> Wt[N][K] bf16 ----
__global__ void wconv_t_kernel(const float* __restrict__ w,
                               unsigned short* __restrict__ wt, int K, int N) {
  int i = blockIdx.x * 256 + threadIdx.x;
  if (i >= K * N) return;
  int n = i / K, k = i % K;
  wt[i] = f2bf(w[(size_t)k * N + n]);
}

// ---------------- fused rel-pos-bias + shift-mask table ---------------------
// bm[w][h][n][m] over padded 64x64; padded entries hold -30000.
__global__ void biasmask_kernel(const float* __restrict__ table, float* __restrict__ bm) {
  int i = blockIdx.x * 256 + threadIdx.x;
  if (i >= NWIN * NH6 * 64 * 64) return;
  int m = i & 63, n = (i >> 6) & 63;
  int wh = i >> 12;                 // w*NH6 + h
  int h = wh % NH6, w = wh / NH6;
  float val = -30000.f;
  if (n < NTOK && m < NTOK) {
    int ny = n / WS7, nx = n % WS7, my = m / WS7, mx = m % WS7;
    int ridx = (ny - my + WS7 - 1) * (2 * WS7 - 1) + (nx - mx + WS7 - 1);
    float bv = table[ridx * NH6 + h];
    int wy = (w >> 3) * WS7, wx = (w & 7) * WS7;
    int rn = 3 * region3(wy + ny) + region3(wx + nx);
    int rm = 3 * region3(wy + my) + region3(wx + mx);
    val = bv + ((rn != rm) ? -100.f : 0.f);
  }
  bm[i] = val;
}

// ---------------- LayerNorm (64 threads per token, 3 channels each) ----------
template <int SHIFTED>
__global__ void ln_kernel(const float* __restrict__ x, const float* __restrict__ g,
                          const float* __restrict__ b, unsigned short* __restrict__ out) {
  __shared__ float red[64];
  __shared__ float red2[64];
  const int tok = blockIdx.x;
  const int tid = threadIdx.x;
  const float* xp = x + (size_t)tok * CC;
  float v0 = xp[tid], v1 = xp[tid + 64], v2 = xp[tid + 128];
  red[tid]  = v0 + v1 + v2;
  red2[tid] = v0 * v0 + v1 * v1 + v2 * v2;
  __syncthreads();
  for (int o = 32; o > 0; o >>= 1) {
    if (tid < o) { red[tid] += red[tid + o]; red2[tid] += red2[tid + o]; }
    __syncthreads();
  }
  float mu  = red[0] * (1.f / CC);
  float var = red2[0] * (1.f / CC) - mu * mu;
  float rs  = rsqrtf(var + 1e-5f);
  size_t dst;
  if (SHIFTED) {
    int bidx = tok / (HH * HH);
    int hw   = tok % (HH * HH);
    int hh = hw / HH, ww = hw % HH;
    int iy = (hh + HH - SS3) % HH, ix = (ww + HH - SS3) % HH;  // roll(-3,-3)
    int w = (iy / WS7) * 8 + (ix / WS7);
    int n = (iy % WS7) * WS7 + (ix % WS7);
    dst = ((size_t)(bidx * NWIN + w) * NTOK + n) * CC;
  } else {
    dst = (size_t)tok * CC;
  }
  float vals[3] = {v0, v1, v2};
  #pragma unroll
  for (int j = 0; j < 3; ++j) {
    int c = tid + 64 * j;
    out[dst + c] = f2bf((vals[j] - mu) * rs * g[c] + b[c]);
  }
}

// ---------------- generic bf16 WMMA GEMM, double-buffered async LDS ---------
//   EPI 0: store bf16 C row-major (stride N)
//   EPI 1: exact GELU then store bf16 (stride N)
//   EPI 2: proj: window-reverse + roll(+3,+3) scatter, outf = resid + val (fp32)
//   EPI 3: fc2:  outf[m*CC+c] += val   (residual add in place)
template <int EPI>
__global__ void __launch_bounds__(256)
gemm_bf16_kernel(const unsigned short* __restrict__ A,
                 const unsigned short* __restrict__ Bt,
                 const float* __restrict__ bias,
                 unsigned short* __restrict__ outbf,
                 float* __restrict__ outf,
                 const float* __restrict__ resid,
                 int K, int N) {
  __shared__ unsigned short Al[2][128 * 32];
  __shared__ unsigned short Bl[2][64 * 32];
  const int tid  = threadIdx.x;
  const int wv   = tid >> 5;        // 8 waves: 4 (rows) x 2 (cols)
  const int lane = tid & 31;
  const int half = lane >> 4;
  const int lr   = lane & 15;
  const int m0 = blockIdx.y * 128;
  const int n0 = blockIdx.x * 64;
  const int RM = (wv >> 1) * 32;
  const int CN = (wv & 1) * 32;

  // per-thread staging chunk coordinates (8 bf16 = 16 B per chunk)
  const int ar0 = tid >> 2, akc = (tid & 3) * 8;   // A rows ar0 and ar0+64
  const int br  = tid >> 2, bkc = (tid & 3) * 8;   // B row (0..63)

  auto stage = [&](int kt, int buf) {
#if HAS_ASYNC_LDS
    __builtin_amdgcn_global_load_async_to_lds_b128(
        (AS1 v4i*)(A + (size_t)(m0 + ar0) * K + kt * 32 + akc),
        (AS3 v4i*)&Al[buf][ar0 * 32 + akc], 0, 0);
    __builtin_amdgcn_global_load_async_to_lds_b128(
        (AS1 v4i*)(A + (size_t)(m0 + ar0 + 64) * K + kt * 32 + akc),
        (AS3 v4i*)&Al[buf][(ar0 + 64) * 32 + akc], 0, 0);
    __builtin_amdgcn_global_load_async_to_lds_b128(
        (AS1 v4i*)(Bt + (size_t)(n0 + br) * K + kt * 32 + bkc),
        (AS3 v4i*)&Bl[buf][br * 32 + bkc], 0, 0);
#else
    *(uint4*)(&Al[buf][ar0 * 32 + akc]) =
        *(const uint4*)(A + (size_t)(m0 + ar0) * K + kt * 32 + akc);
    *(uint4*)(&Al[buf][(ar0 + 64) * 32 + akc]) =
        *(const uint4*)(A + (size_t)(m0 + ar0 + 64) * K + kt * 32 + akc);
    *(uint4*)(&Bl[buf][br * 32 + bkc]) =
        *(const uint4*)(Bt + (size_t)(n0 + br) * K + kt * 32 + bkc);
#endif
  };

  v8f acc[2][2];
  #pragma unroll
  for (int i = 0; i < 2; ++i)
    #pragma unroll
    for (int j = 0; j < 2; ++j) acc[i][j] = vzero8();

  const int nkt = K >> 5;
  stage(0, 0);
  async_wait0();
  __syncthreads();

  for (int kt = 0; kt < nkt; ++kt) {
    const int cur = kt & 1;
    if (kt + 1 < nkt) stage(kt + 1, cur ^ 1);   // overlap next tile with compute
    #pragma unroll
    for (int i = 0; i < 2; ++i) {
      Frag16 fa;
      int arow = RM + i * 16 + lr;
      fa.q[0] = *(const uint4*)(&Al[cur][arow * 32 + 8 * half]);
      fa.q[1] = *(const uint4*)(&Al[cur][arow * 32 + 16 + 8 * half]);
      #pragma unroll
      for (int j = 0; j < 2; ++j) {
        Frag16 fb;
        int bcol = CN + j * 16 + lr;
        fb.q[0] = *(const uint4*)(&Bl[cur][bcol * 32 + 8 * half]);
        fb.q[1] = *(const uint4*)(&Bl[cur][bcol * 32 + 16 + 8 * half]);
        acc[i][j] = __builtin_amdgcn_wmma_f32_16x16x32_bf16(
            false, fa.v, false, fb.v, (short)0, acc[i][j], false, false);
      }
    }
    if (kt + 1 < nkt) {
      async_wait0();
      __syncthreads();
    }
  }

  // epilogue
  #pragma unroll
  for (int i = 0; i < 2; ++i) {
    #pragma unroll
    for (int j = 0; j < 2; ++j) {
      #pragma unroll
      for (int e = 0; e < 8; ++e) {
        int m = m0 + RM + i * 16 + e + 8 * half;
        int c = n0 + CN + j * 16 + lr;
        float v = acc[i][j][e] + bias[c];
        if (EPI == 0) {
          outbf[(size_t)m * N + c] = f2bf(v);
        } else if (EPI == 1) {
          float gl = 0.5f * v * (1.f + erff(v * 0.70710678118654752f));
          outbf[(size_t)m * N + c] = f2bf(gl);
        } else if (EPI == 2) {
          int bidx = m / (NWIN * NTOK);
          int r = m % (NWIN * NTOK);
          int w = r / NTOK, n = r % NTOK;
          int iy = (w >> 3) * WS7 + n / WS7;
          int ix = (w & 7) * WS7 + n % WS7;
          int hh = (iy + SS3) % HH, ww = (ix + SS3) % HH;  // roll(+3,+3)
          size_t o = ((size_t)(bidx * HH + hh) * HH + ww) * CC + c;
          outf[o] = resid[o] + v;
        } else {
          size_t o = (size_t)m * CC + c;
          outf[o] = outf[o] + v;
        }
      }
    }
  }
}

// ---------------- windowed attention (one block per window*head) ------------
__global__ void __launch_bounds__(128)
attn_kernel(const unsigned short* __restrict__ qkv,   // [Bw*49][576] bf16
            const float* __restrict__ bm,             // [NWIN][NH][64][64] fused bias+mask
            unsigned short* __restrict__ obf) {       // [Bw*49][192] bf16
  __shared__ unsigned short Ql[64 * 32];
  __shared__ unsigned short Kl[64 * 32];
  __shared__ unsigned short Vt[32 * 64];   // transposed: [d][m]
  __shared__ float          Sl[64 * 64];
  __shared__ unsigned short Pl[64 * 64];

  const int blk  = blockIdx.x;             // (b*NWIN + w)*NH + h
  const int h    = blk % NH6;
  const int bw   = blk / NH6;              // b*64 + w
  const int w    = bw % NWIN;
  const int tid  = threadIdx.x;
  const int wv   = tid >> 5;               // 4 waves, one 16-row band each
  const int lane = tid & 31;
  const int half = lane >> 4;
  const int lr   = lane & 15;

  const float* bmp = bm + ((size_t)(w * NH6 + h) << 12);

  // ---- load Q,K rows (vectorized), zero-pad rows 49..63 ----
  #pragma unroll
  for (int it = 0; it < 2; ++it) {
    int c = tid + 128 * it;                // 256 chunk jobs (64 rows x 4 chunks)
    int row = c >> 2, kc = (c & 3) * 8;
    uint4 zq; zq.x = zq.y = zq.z = zq.w = 0u;
    uint4 qv = zq, kv = zq;
    if (row < NTOK) {
      const unsigned short* base = qkv + ((size_t)bw * NTOK + row) * (3 * CC) + h * DH32 + kc;
      qv = *(const uint4*)(base);
      kv = *(const uint4*)(base + CC);
    }
    *(uint4*)(&Ql[row * 32 + kc]) = qv;
    *(uint4*)(&Kl[row * 32 + kc]) = kv;
  }
  // ---- load V transposed: Vt[d][m] ----
  for (int c = tid; c < 2048; c += 128) {
    int d = c >> 6, m = c & 63;
    unsigned short v = 0;
    if (m < NTOK)
      v = qkv[((size_t)bw * NTOK + m) * (3 * CC) + 2 * CC + h * DH32 + d];
    Vt[d * 64 + m] = v;
  }
  __syncthreads();

  // ---- S = Q * K^T (rows 16*wv..16*wv+15, all 64 cols, single k=32 step) ----
  v8f s[4];
  #pragma unroll
  for (int t = 0; t < 4; ++t) s[t] = vzero8();
  {
    Frag16 fa;
    int arow = wv * 16 + lr;
    fa.q[0] = *(const uint4*)(&Ql[arow * 32 + 8 * half]);
    fa.q[1] = *(const uint4*)(&Ql[arow * 32 + 16 + 8 * half]);
    #pragma unroll
    for (int t = 0; t < 4; ++t) {
      Frag16 fb;
      int bcol = t * 16 + lr;
      fb.q[0] = *(const uint4*)(&Kl[bcol * 32 + 8 * half]);
      fb.q[1] = *(const uint4*)(&Kl[bcol * 32 + 16 + 8 * half]);
      s[t] = __builtin_amdgcn_wmma_f32_16x16x32_bf16(
          false, fa.v, false, fb.v, (short)0, s[t], false, false);
    }
  }

  // ---- scale + precomputed (bias+mask) -> Sl -------------------------------
  #pragma unroll
  for (int t = 0; t < 4; ++t) {
    #pragma unroll
    for (int e = 0; e < 8; ++e) {
      int n = wv * 16 + e + 8 * half;     // query index
      int m = t * 16 + lr;                // key index
      Sl[n * 64 + m] = s[t][e] * QKSCALE + bmp[n * 64 + m];
    }
  }
  __syncthreads();

  // ---- softmax per row (one thread per row, rows 0..63) ----
  if (tid < 64) {
    int n = tid;
    float mx = -1e30f;
    for (int m = 0; m < NTOK; ++m) mx = fmaxf(mx, Sl[n * 64 + m]);
    float sum = 0.f;
    for (int m = 0; m < NTOK; ++m) sum += __expf(Sl[n * 64 + m] - mx);
    float inv = 1.f / sum;
    for (int m = 0; m < 64; ++m) {
      float p = (m < NTOK) ? __expf(Sl[n * 64 + m] - mx) * inv : 0.f;
      Pl[n * 64 + m] = f2bf(p);
    }
  }
  __syncthreads();

  // ---- O = P * V : 2 k-steps over m (64), 2 col tiles over d (32) ----
  v8f o[2]; o[0] = vzero8(); o[1] = vzero8();
  #pragma unroll
  for (int ks = 0; ks < 2; ++ks) {
    Frag16 fa;
    int arow = wv * 16 + lr;
    fa.q[0] = *(const uint4*)(&Pl[arow * 64 + ks * 32 + 8 * half]);
    fa.q[1] = *(const uint4*)(&Pl[arow * 64 + ks * 32 + 16 + 8 * half]);
    #pragma unroll
    for (int t = 0; t < 2; ++t) {
      Frag16 fb;
      int dcol = t * 16 + lr;
      fb.q[0] = *(const uint4*)(&Vt[dcol * 64 + ks * 32 + 8 * half]);
      fb.q[1] = *(const uint4*)(&Vt[dcol * 64 + ks * 32 + 16 + 8 * half]);
      o[t] = __builtin_amdgcn_wmma_f32_16x16x32_bf16(
          false, fa.v, false, fb.v, (short)0, o[t], false, false);
    }
  }
  #pragma unroll
  for (int t = 0; t < 2; ++t) {
    #pragma unroll
    for (int e = 0; e < 8; ++e) {
      int n = wv * 16 + e + 8 * half;
      int d = t * 16 + lr;
      if (n < NTOK)
        obf[((size_t)bw * NTOK + n) * CC + h * DH32 + d] = f2bf(o[t][e]);
    }
  }
}

// ---------------- host-side orchestration -----------------------------------
extern "C" void kernel_launch(void* const* d_in, const int* in_sizes, int n_in,
                              void* d_out, int out_size, void* d_ws, size_t ws_size,
                              hipStream_t stream) {
  const float* x        = (const float*)d_in[0];
  const float* norm1_g  = (const float*)d_in[1];
  const float* norm1_b  = (const float*)d_in[2];
  const float* qkv_w    = (const float*)d_in[3];
  const float* qkv_b    = (const float*)d_in[4];
  const float* rel_bias = (const float*)d_in[5];
  const float* proj_w   = (const float*)d_in[6];
  const float* proj_b   = (const float*)d_in[7];
  const float* norm2_g  = (const float*)d_in[8];
  const float* norm2_b  = (const float*)d_in[9];
  const float* fc1_w    = (const float*)d_in[10];
  const float* fc1_b    = (const float*)d_in[11];
  const float* fc2_w    = (const float*)d_in[12];
  const float* fc2_b    = (const float*)d_in[13];
  float* out = (float*)d_out;

  unsigned char* ws = (unsigned char*)d_ws;
  size_t o_wqkv = 0;                                   // 576*192 bf16
  size_t o_wprj = o_wqkv + (size_t)576 * 192 * 2;      // 192*192 bf16
  size_t o_wfc1 = o_wprj + (size_t)192 * 192 * 2;      // 768*192 bf16
  size_t o_wfc2 = o_wfc1 + (size_t)768 * 192 * 2;      // 192*768 bf16
  size_t o_bm   = o_wfc2 + (size_t)192 * 768 * 2;      // NWIN*NH*64*64 f32
  size_t o_xw   = o_bm   + (size_t)NWIN * NH6 * 64 * 64 * 4; // MTOT*192 bf16 (xw, then xn2)
  size_t o_qkv  = o_xw   + (size_t)MTOT * CC * 2;      // MTOT*576 bf16 (qkv; start of h)
  size_t o_obf  = o_qkv  + (size_t)MTOT * 3 * CC * 2;  // MTOT*192 bf16 (attn out)
  size_t o_h    = o_qkv;                               // MTOT*768 bf16 reuses qkv+obf

  unsigned short* wqkv = (unsigned short*)(ws + o_wqkv);
  unsigned short* wprj = (unsigned short*)(ws + o_wprj);
  unsigned short* wfc1 = (unsigned short*)(ws + o_wfc1);
  unsigned short* wfc2 = (unsigned short*)(ws + o_wfc2);
  float*          bm   = (float*)(ws + o_bm);
  unsigned short* xw   = (unsigned short*)(ws + o_xw);
  unsigned short* qkvb = (unsigned short*)(ws + o_qkv);
  unsigned short* obf  = (unsigned short*)(ws + o_obf);
  unsigned short* hbuf = (unsigned short*)(ws + o_h);

  // 1) weight convert + transpose, fused bias+mask table
  wconv_t_kernel<<<(192 * 576 + 255) / 256, 256, 0, stream>>>(qkv_w, wqkv, 192, 576);
  wconv_t_kernel<<<(192 * 192 + 255) / 256, 256, 0, stream>>>(proj_w, wprj, 192, 192);
  wconv_t_kernel<<<(192 * 768 + 255) / 256, 256, 0, stream>>>(fc1_w, wfc1, 192, 768);
  wconv_t_kernel<<<(768 * 192 + 255) / 256, 256, 0, stream>>>(fc2_w, wfc2, 768, 192);
  biasmask_kernel<<<(NWIN * NH6 * 64 * 64) / 256, 256, 0, stream>>>(rel_bias, bm);

  // 2) LN1 + roll(-3,-3) + window partition -> bf16
  ln_kernel<1><<<MTOT, 64, 0, stream>>>(x, norm1_g, norm1_b, xw);

  // 3) QKV GEMM: (MTOT,192)x(192,576)
  gemm_bf16_kernel<0><<<dim3(576 / 64, MTOT / 128), 256, 0, stream>>>(
      xw, wqkv, qkv_b, qkvb, nullptr, nullptr, 192, 576);

  // 4) windowed attention
  attn_kernel<<<BB * NWIN * NH6, 128, 0, stream>>>(qkvb, bm, obf);

  // 5) proj GEMM + window reverse + roll(+3,+3) + residual -> d_out (x1, fp32)
  gemm_bf16_kernel<2><<<dim3(192 / 64, MTOT / 128), 256, 0, stream>>>(
      obf, wprj, proj_b, nullptr, out, x, 192, 192);

  // 6) LN2 (reads x1 from d_out) -> bf16 (reuses xw region)
  ln_kernel<0><<<MTOT, 64, 0, stream>>>(out, norm2_g, norm2_b, xw);

  // 7) fc1 + exact GELU: (MTOT,192)x(192,768)
  gemm_bf16_kernel<1><<<dim3(768 / 64, MTOT / 128), 256, 0, stream>>>(
      xw, wfc1, fc1_b, hbuf, nullptr, nullptr, 192, 768);

  // 8) fc2 + residual add into d_out: (MTOT,768)x(768,192)
  gemm_bf16_kernel<3><<<dim3(192 / 64, MTOT / 128), 256, 0, stream>>>(
      hbuf, wfc2, fc2_b, nullptr, out, nullptr, 768, 192);

  (void)in_sizes; (void)n_in; (void)out_size; (void)ws_size;
}
